// SwinTransformerBlock_80900003988047
// MI455X (gfx1250) — matrix operand order repaired and checked by reading, not compiled, not run on hardware
//
#include <hip/hip_runtime.h>
#include <hip/hip_bf16.h>
#include <stdint.h>

// ---------------- problem constants ----------------
constexpr int NB     = 16;            // batch
constexpr int IH     = 64;            // image H
constexpr int IW     = 128;           // image W
constexpr int CD     = 512;           // channels
constexpr int NHEAD  = 8;
constexpr int HDIM   = 64;
constexpr int NTOK   = 32;            // tokens per window (4x8)
constexpr int HIDDEN = 2048;
constexpr int NWIN   = 4096;          // total windows = 16 * 16 * 16
constexpr int MROWS  = NWIN * NTOK;   // 131072 GEMM rows
constexpr float ATT_SCALE = 0.125f;   // HD^-0.5
constexpr float LNEPS = 1e-5f;

// ---------------- vector types ----------------
typedef __attribute__((ext_vector_type(16))) __bf16 v16bf;
typedef __attribute__((ext_vector_type(8)))  __bf16 v8bf;
typedef __attribute__((ext_vector_type(4)))  __bf16 v4bf;
typedef __attribute__((ext_vector_type(8)))  float  v8f;
typedef __attribute__((ext_vector_type(4)))  float  v4f;

// ---------------- bf16 helpers (RNE) ----------------
static __device__ inline __bf16 f2bf(float f) {
  union { float f; unsigned u; } x; x.f = f;
  unsigned r = x.u + 0x7fffu + ((x.u >> 16) & 1u);
  unsigned short h = (unsigned short)(r >> 16);
  __bf16 o; __builtin_memcpy(&o, &h, 2); return o;
}
static __device__ inline float bf2f(__bf16 b) {
  unsigned short h; __builtin_memcpy(&h, &b, 2);
  union { unsigned u; float f; } x; x.u = ((unsigned)h) << 16; return x.f;
}
static __device__ inline v8f vzero8() {
  v8f z;
#pragma unroll
  for (int i = 0; i < 8; ++i) z[i] = 0.f;
  return z;
}

// ---------------- WMMA fragment loads ----------------
// A (16-bit 16x32, ISA 7.12.2): lane L: M = L&15, khalf base = (L<16)?0:8,
//   a[j] = A[M][base+j] (j<8) ; a[j] = A[M][16+base+j-8] (j>=8)
static __device__ inline v16bf load_frag16(const __bf16* p0, const __bf16* p1) {
  v8bf lo = *(const v8bf*)p0;
  v8bf hi = *(const v8bf*)p1;
  v16bf r;
#pragma unroll
  for (int i = 0; i < 8; ++i) { r[i] = lo[i]; r[i + 8] = hi[i]; }
  return r;
}
static __device__ inline v16bf load_a_frag(const __bf16* tile, int ldk) {
  int lane = threadIdx.x & 31;
  int m = lane & 15;
  int off = (lane < 16) ? 0 : 8;
  const __bf16* p = tile + (size_t)m * ldk + off;
  return load_frag16(p, p + 16);
}
// B (16-bit 32x16): lane L: N = L&15, K base = (L<16)?0:8, b[j] = B[base+j][N] (j<8), b[j] = B[16+base+j-8][N] (j>=8).
// With B[k][n] == Wt[n][k] (weights stored (N,K) row-major) this is two
// 8-wide contiguous bf16 chunks per lane.
static __device__ inline v16bf load_b_frag(const __bf16* wt, int ldk) {
  int lane = threadIdx.x & 31;
  int n = lane & 15;
  int off = (lane < 16) ? 0 : 8;
  const __bf16* p = wt + (size_t)n * ldk + off;
  return load_frag16(p, p + 16);
}
static __device__ inline v8f wmma_bf16(v16bf a, v16bf b, v8f c) {
  return __builtin_amdgcn_wmma_f32_16x16x32_bf16(false, a, false, b, (short)0, c,
                                                 false, false);
}

// window row -> (B,HW) row in x / y coords (includes the +-(2,4) roll)
static __device__ inline int win_row_to_xrow(int r) {
  int w = r >> 5, n = r & 31;
  int b = w >> 8, wim = w & 255;
  int hb = wim >> 4, wb = wim & 15;
  int wh = n >> 3, wwi = n & 7;
  int h  = (hb * 4 + wh + 2) & 63;
  int wc = (wb * 8 + wwi + 4) & 127;
  return (b << 13) + (h << 7) + wc;
}

// ---------------- kernel 0: f32 -> bf16 convert ----------------
__global__ __launch_bounds__(256) void k_cvt(const float* __restrict__ src,
                                             __bf16* __restrict__ dst, int n4) {
  int i = blockIdx.x * 256 + threadIdx.x;
  if (i >= n4) return;
  v4f f = *(const v4f*)(src + (size_t)i * 4);
  v4bf o;
#pragma unroll
  for (int j = 0; j < 4; ++j) o[j] = f2bf(f[j]);
  *(v4bf*)(dst + (size_t)i * 4) = o;
}

// ---------------- kernel 1: shifted-window gather, fp32 -> bf16 ----------------
__global__ __launch_bounds__(256) void k_gather(const float* __restrict__ x,
                                                __bf16* __restrict__ Xw) {
  int idx = blockIdx.x * 256 + threadIdx.x;     // MROWS*128 threads
  int r = idx >> 7;
  int c = (idx & 127) << 2;
  int xr = win_row_to_xrow(r);
  v4f f = *(const v4f*)(x + (size_t)xr * CD + c);
  v4bf o;
#pragma unroll
  for (int j = 0; j < 4; ++j) o[j] = f2bf(f[j]);
  *(v4bf*)(Xw + (size_t)r * CD + c) = o;
}

// ---------------- kernel 2: QKV GEMM (M=131072,K=512,N=1536) ----------------
// grid (4096, 3); blockIdx.y selects q/k/v slice; 8 waves x 64 cols each.
__global__ __launch_bounds__(256) void k_qkv(const __bf16* __restrict__ Xw,
                                             const __bf16* __restrict__ wq,
                                             const float* __restrict__ qkv_b,
                                             __bf16* __restrict__ q,
                                             __bf16* __restrict__ k,
                                             __bf16* __restrict__ v) {
  int wave = threadIdx.x >> 5, lane = threadIdx.x & 31;
  int hi16 = lane >> 4, l15 = lane & 15;
  int r0 = blockIdx.x * 32;
  int mat = blockIdx.y;
  const __bf16* W = wq + (size_t)mat * CD * CD;
  v8f acc[2][4];
#pragma unroll
  for (int a = 0; a < 2; ++a)
#pragma unroll
    for (int b = 0; b < 4; ++b) acc[a][b] = vzero8();

  for (int kc = 0; kc < CD; kc += 32) {
    __builtin_prefetch(Xw + (size_t)r0 * CD + kc + 64, 0, 0);
    v16bf a0 = load_a_frag(Xw + (size_t)r0 * CD + kc, CD);
    v16bf a1 = load_a_frag(Xw + (size_t)(r0 + 16) * CD + kc, CD);
#pragma unroll
    for (int t = 0; t < 4; ++t) {
      v16bf b = load_b_frag(W + (size_t)(wave * 64 + t * 16) * CD + kc, CD);
      acc[0][t] = wmma_bf16(a0, b, acc[0][t]);
      acc[1][t] = wmma_bf16(a1, b, acc[1][t]);
    }
  }
  __bf16* out = (mat == 0) ? q : (mat == 1) ? k : v;
  int head = wave;
#pragma unroll
  for (int mt = 0; mt < 2; ++mt)
#pragma unroll
    for (int t = 0; t < 4; ++t)
#pragma unroll
      for (int i = 0; i < 8; ++i) {
        int m = r0 + mt * 16 + hi16 * 8 + i;
        int d = t * 16 + l15;
        float val = acc[mt][t][i] + qkv_b[mat * CD + wave * 64 + d];
        int wn = m >> 5, n = m & 31;
        out[(((size_t)wn * NHEAD + head) * NTOK + n) * HDIM + d] = f2bf(val);
      }
}

// ---------------- kernel 3: 2D RoPE on q and k ----------------
__global__ __launch_bounds__(256) void k_rope(__bf16* __restrict__ q,
                                              __bf16* __restrict__ k) {
  __bf16* t = blockIdx.y ? k : q;
  int p = blockIdx.x * 256 + threadIdx.x;       // pair index
  int d2 = p & 31;
  int n  = (p >> 5) & 31;
  size_t base = ((size_t)(p >> 10)) * 2048 + (size_t)n * 64 + (size_t)d2 * 2;
  int pos, i;
  if (d2 < 16) { i = d2;      pos = n >> 3; }   // h-half uses wh
  else         { i = d2 - 16; pos = n & 7;  }   // w-half uses ww
  float inv = __expf(-0.57564627325f * (float)i);   // 10000^(-i/16)
  float ang = (float)pos * inv;
  float cs = __cosf(ang), sn = __sinf(ang);
  float e = bf2f(t[base]), o = bf2f(t[base + 1]);
  t[base]     = f2bf(e * cs - o * sn);
  t[base + 1] = f2bf(o * cs + e * sn);
}

// ---------------- kernel 4: windowed attention ----------------
// grid 4096 (one window); 4 waves, each wave does 2 heads sequentially.
__global__ __launch_bounds__(128) void k_attn(const __bf16* __restrict__ q,
                                              const __bf16* __restrict__ k,
                                              const __bf16* __restrict__ v,
                                              const float* __restrict__ mask,
                                              __bf16* __restrict__ att) {
  __shared__ float  S[4][32][33];
  __shared__ __bf16 P[4][32][40];
  __shared__ __bf16 VT[4][64][40];
  int wave = threadIdx.x >> 5, lane = threadIdx.x & 31;
  int hi16 = lane >> 4, l15 = lane & 15;
  int w = blockIdx.x, wim = w & 255;

  for (int hh = 0; hh < 2; ++hh) {
    int head = wave * 2 + hh;
    const __bf16* qp = q + ((size_t)(w * NHEAD + head)) * (NTOK * HDIM);
    const __bf16* kp = k + ((size_t)(w * NHEAD + head)) * (NTOK * HDIM);
    const __bf16* vp = v + ((size_t)(w * NHEAD + head)) * (NTOK * HDIM);

    // stage V transposed (lane = token)
#pragma unroll
    for (int dv = 0; dv < 8; ++dv) {
      v8bf chnk = *(const v8bf*)(vp + (size_t)lane * HDIM + dv * 8);
#pragma unroll
      for (int e = 0; e < 8; ++e) VT[wave][dv * 8 + e][lane] = chnk[e];
    }

    // S = q . k^T  (32x32, K=64)
    v8f sacc[2][2];
#pragma unroll
    for (int a = 0; a < 2; ++a)
#pragma unroll
      for (int b = 0; b < 2; ++b) sacc[a][b] = vzero8();
#pragma unroll
    for (int kt = 0; kt < 2; ++kt) {
      int kc = kt * 32;
      v16bf a0 = load_a_frag(qp + kc, HDIM);
      v16bf a1 = load_a_frag(qp + 16 * HDIM + kc, HDIM);
#pragma unroll
      for (int nt = 0; nt < 2; ++nt) {
        v16bf b = load_b_frag(kp + (size_t)(nt * 16) * HDIM + kc, HDIM);
        sacc[0][nt] = wmma_bf16(a0, b, sacc[0][nt]);
        sacc[1][nt] = wmma_bf16(a1, b, sacc[1][nt]);
      }
    }
#pragma unroll
    for (int mt = 0; mt < 2; ++mt)
#pragma unroll
      for (int nt = 0; nt < 2; ++nt)
#pragma unroll
        for (int i = 0; i < 8; ++i) {
          int row = mt * 16 + hi16 * 8 + i;
          int col = nt * 16 + l15;
          S[wave][row][col] = sacc[mt][nt][i] * ATT_SCALE +
                              mask[((size_t)wim * 32 + row) * 32 + col];
        }
    __syncthreads();

    // softmax: one row per lane
    {
      float rv[32];
      float mx = -1e30f;
#pragma unroll
      for (int j = 0; j < 32; ++j) { rv[j] = S[wave][lane][j]; mx = fmaxf(mx, rv[j]); }
      float sum = 0.f;
#pragma unroll
      for (int j = 0; j < 32; ++j) { rv[j] = __expf(rv[j] - mx); sum += rv[j]; }
      float r = 1.f / sum;
#pragma unroll
      for (int j = 0; j < 32; ++j) P[wave][lane][j] = f2bf(rv[j] * r);
    }
    __syncthreads();

    // O = P . V  (32x64, K=32 -> single WMMA k-step)
    v16bf a0 = load_a_frag(&P[wave][0][0], 40);
    v16bf a1 = load_a_frag(&P[wave][16][0], 40);
    v8f oacc[2][4];
#pragma unroll
    for (int a = 0; a < 2; ++a)
#pragma unroll
      for (int b = 0; b < 4; ++b) oacc[a][b] = vzero8();
#pragma unroll
    for (int t = 0; t < 4; ++t) {
      v16bf b = load_b_frag(&VT[wave][t * 16][0], 40);
      oacc[0][t] = wmma_bf16(a0, b, oacc[0][t]);
      oacc[1][t] = wmma_bf16(a1, b, oacc[1][t]);
    }
#pragma unroll
    for (int mt = 0; mt < 2; ++mt)
#pragma unroll
      for (int t = 0; t < 4; ++t)
#pragma unroll
        for (int i = 0; i < 8; ++i) {
          int m = mt * 16 + hi16 * 8 + i;
          int col = head * 64 + t * 16 + l15;
          att[((size_t)w * NTOK + m) * CD + col] = f2bf(oacc[mt][t][i]);
        }
    __syncthreads();   // LDS reuse for next head
  }
}

// ---------------- kernel 5: proj GEMM + reverse roll + residual + LN1 --------
__global__ __launch_bounds__(256) void k_proj_ln1(const __bf16* __restrict__ att,
                                                  const __bf16* __restrict__ wp,
                                                  const float* __restrict__ proj_b,
                                                  const float* __restrict__ x,
                                                  const float* __restrict__ g,
                                                  const float* __restrict__ bt,
                                                  float* __restrict__ yf,
                                                  __bf16* __restrict__ ybf) {
  __shared__ float rsum[32], rsq[32];
  int wave = threadIdx.x >> 5, lane = threadIdx.x & 31;
  int hi16 = lane >> 4, l15 = lane & 15;
  int r0 = blockIdx.x * 32;
  v8f acc[2][4];
#pragma unroll
  for (int a = 0; a < 2; ++a)
#pragma unroll
    for (int b = 0; b < 4; ++b) acc[a][b] = vzero8();

  for (int kc = 0; kc < CD; kc += 32) {
    __builtin_prefetch(att + (size_t)r0 * CD + kc + 64, 0, 0);
    v16bf a0 = load_a_frag(att + (size_t)r0 * CD + kc, CD);
    v16bf a1 = load_a_frag(att + (size_t)(r0 + 16) * CD + kc, CD);
#pragma unroll
    for (int t = 0; t < 4; ++t) {
      v16bf b = load_b_frag(wp + (size_t)(wave * 64 + t * 16) * CD + kc, CD);
      acc[0][t] = wmma_bf16(a0, b, acc[0][t]);
      acc[1][t] = wmma_bf16(a1, b, acc[1][t]);
    }
  }
  if (threadIdx.x < 32) { rsum[threadIdx.x] = 0.f; rsq[threadIdx.x] = 0.f; }
  __syncthreads();

#pragma unroll
  for (int mt = 0; mt < 2; ++mt)
#pragma unroll
    for (int i = 0; i < 8; ++i) {
      int ml = mt * 16 + hi16 * 8 + i;
      int xr = win_row_to_xrow(r0 + ml);
      float s = 0.f, ss = 0.f;
#pragma unroll
      for (int t = 0; t < 4; ++t) {
        int col = wave * 64 + t * 16 + l15;
        float val = acc[mt][t][i] + proj_b[col] + x[(size_t)xr * CD + col];
        acc[mt][t][i] = val;
        s += val; ss += val * val;
      }
      atomicAdd(&rsum[ml], s);
      atomicAdd(&rsq[ml], ss);
    }
  __syncthreads();

#pragma unroll
  for (int mt = 0; mt < 2; ++mt)
#pragma unroll
    for (int i = 0; i < 8; ++i) {
      int ml = mt * 16 + hi16 * 8 + i;
      int xr = win_row_to_xrow(r0 + ml);
      float mu = rsum[ml] * (1.f / 512.f);
      float var = rsq[ml] * (1.f / 512.f) - mu * mu;
      float rs = rsqrtf(var + LNEPS);
#pragma unroll
      for (int t = 0; t < 4; ++t) {
        int col = wave * 64 + t * 16 + l15;
        float val = (acc[mt][t][i] - mu) * rs * g[col] + bt[col];
        yf[(size_t)xr * CD + col] = val;
        ybf[(size_t)xr * CD + col] = f2bf(val);
      }
    }
}

// ---------------- kernel 6: fused MLP (fc1+relu+fc2) + residual + LN2 --------
// grid 4096 (32 rows each); hidden never hits HBM (64-wide chunks via LDS).
__global__ __launch_bounds__(256) void k_mlp_ln2(const __bf16* __restrict__ ybf,
                                                 const float* __restrict__ yf,
                                                 const __bf16* __restrict__ w1,
                                                 const float* __restrict__ b1,
                                                 const __bf16* __restrict__ w2,
                                                 const float* __restrict__ b2,
                                                 const float* __restrict__ g,
                                                 const float* __restrict__ bt,
                                                 float* __restrict__ out) {
  __shared__ __bf16 Hc[32][72];
  __shared__ float rsum[32], rsq[32];
  int wave = threadIdx.x >> 5, lane = threadIdx.x & 31;
  int hi16 = lane >> 4, l15 = lane & 15;
  int r0 = blockIdx.x * 32;
  int mt1 = wave & 1, nt1 = wave >> 1;   // stage-1 fragment ownership

  v8f acc[2][4];
#pragma unroll
  for (int a = 0; a < 2; ++a)
#pragma unroll
    for (int b = 0; b < 4; ++b) acc[a][b] = vzero8();

  for (int ch = 0; ch < HIDDEN / 64; ++ch) {
    // stage 1: 32x64 hidden chunk, K=512
    v8f hacc = vzero8();
    int hc0 = ch * 64 + nt1 * 16;
    for (int kc = 0; kc < CD; kc += 32) {
      __builtin_prefetch(w1 + (size_t)hc0 * CD + kc + 64, 0, 0);
      v16bf a = load_a_frag(ybf + (size_t)(r0 + mt1 * 16) * CD + kc, CD);
      v16bf b = load_b_frag(w1 + (size_t)hc0 * CD + kc, CD);
      hacc = wmma_bf16(a, b, hacc);
    }
#pragma unroll
    for (int i = 0; i < 8; ++i) {
      int hr = mt1 * 16 + hi16 * 8 + i;
      float valh = hacc[i] + b1[hc0 + l15];
      Hc[hr][nt1 * 16 + l15] = f2bf(fmaxf(valh, 0.f));
    }
    __syncthreads();

    // stage 2: accumulate 32x512 output, K=64 (2 steps)
#pragma unroll
    for (int kt = 0; kt < 2; ++kt) {
      int kcc = kt * 32;
      v16bf a0 = load_a_frag(&Hc[0][kcc], 72);
      v16bf a1 = load_a_frag(&Hc[16][kcc], 72);
#pragma unroll
      for (int t = 0; t < 4; ++t) {
        v16bf b = load_b_frag(w2 + (size_t)(wave * 64 + t * 16) * HIDDEN +
                                  ch * 64 + kcc, HIDDEN);
        acc[0][t] = wmma_bf16(a0, b, acc[0][t]);
        acc[1][t] = wmma_bf16(a1, b, acc[1][t]);
      }
    }
    __syncthreads();   // protect Hc before next chunk
  }

  // epilogue: bias + residual + LN2
  if (threadIdx.x < 32) { rsum[threadIdx.x] = 0.f; rsq[threadIdx.x] = 0.f; }
  __syncthreads();
#pragma unroll
  for (int mt = 0; mt < 2; ++mt)
#pragma unroll
    for (int i = 0; i < 8; ++i) {
      int ml = mt * 16 + hi16 * 8 + i;
      float s = 0.f, ss = 0.f;
#pragma unroll
      for (int t = 0; t < 4; ++t) {
        int col = wave * 64 + t * 16 + l15;
        float val = acc[mt][t][i] + b2[col] + yf[(size_t)(r0 + ml) * CD + col];
        acc[mt][t][i] = val;
        s += val; ss += val * val;
      }
      atomicAdd(&rsum[ml], s);
      atomicAdd(&rsq[ml], ss);
    }
  __syncthreads();
#pragma unroll
  for (int mt = 0; mt < 2; ++mt)
#pragma unroll
    for (int i = 0; i < 8; ++i) {
      int ml = mt * 16 + hi16 * 8 + i;
      float mu = rsum[ml] * (1.f / 512.f);
      float var = rsq[ml] * (1.f / 512.f) - mu * mu;
      float rs = rsqrtf(var + LNEPS);
#pragma unroll
      for (int t = 0; t < 4; ++t) {
        int col = wave * 64 + t * 16 + l15;
        out[(size_t)(r0 + ml) * CD + col] = (acc[mt][t][i] - mu) * rs * g[col] + bt[col];
      }
    }
}

// ---------------- host launcher ----------------
extern "C" void kernel_launch(void* const* d_in, const int* in_sizes, int n_in,
                              void* d_out, int out_size, void* d_ws, size_t ws_size,
                              hipStream_t stream) {
  const float* x      = (const float*)d_in[0];
  const float* mask   = (const float*)d_in[1];
  const float* qkv_w  = (const float*)d_in[2];
  const float* qkv_b  = (const float*)d_in[3];
  const float* proj_w = (const float*)d_in[4];
  const float* proj_b = (const float*)d_in[5];
  const float* ln1_g  = (const float*)d_in[6];
  const float* ln1_b  = (const float*)d_in[7];
  const float* ln2_g  = (const float*)d_in[8];
  const float* ln2_b  = (const float*)d_in[9];
  const float* fc1_w  = (const float*)d_in[10];
  const float* fc1_b  = (const float*)d_in[11];
  const float* fc2_w  = (const float*)d_in[12];
  const float* fc2_b  = (const float*)d_in[13];

  char* ws = (char*)d_ws;
  const size_t ACT = (size_t)MROWS * CD * sizeof(__bf16);   // 134,217,728 B
  __bf16* wq = (__bf16*)(ws + 0);
  __bf16* wp = (__bf16*)(ws + 1572864);
  __bf16* w1 = (__bf16*)(ws + 2097152);
  __bf16* w2 = (__bf16*)(ws + 4194304);
  __bf16* Xw = (__bf16*)(ws + 8388608);
  __bf16* qb = (__bf16*)(ws + 8388608 + ACT);
  __bf16* kb = (__bf16*)(ws + 8388608 + 2 * ACT);
  __bf16* vb = (__bf16*)(ws + 8388608 + 3 * ACT);
  __bf16* att = Xw;              // Xw dead after QKV
  __bf16* ybf = qb;              // q dead after attention
  float*  yf  = (float*)kb;      // k+v dead after attention (spans 2*ACT)

  // 0) weights -> bf16 (stay resident in L2 for every GEMM)
  k_cvt<<<768,  256, 0, stream>>>(qkv_w, wq, 1536 * 512 / 4);
  k_cvt<<<256,  256, 0, stream>>>(proj_w, wp, 512 * 512 / 4);
  k_cvt<<<1024, 256, 0, stream>>>(fc1_w, w1, 2048 * 512 / 4);
  k_cvt<<<1024, 256, 0, stream>>>(fc2_w, w2, 512 * 2048 / 4);
  // 1) shifted-window gather + cast
  k_gather<<<65536, 256, 0, stream>>>(x, Xw);
  // 2) QKV GEMM
  k_qkv<<<dim3(NWIN, 3), 256, 0, stream>>>(Xw, wq, qkv_b, qb, kb, vb);
  // 3) RoPE on q,k
  k_rope<<<dim3(131072, 2), 256, 0, stream>>>(qb, kb);
  // 4) attention per window
  k_attn<<<NWIN, 128, 0, stream>>>(qb, kb, vb, mask, att);
  // 5) proj + reverse roll + residual + LN1
  k_proj_ln1<<<NWIN, 256, 0, stream>>>(att, wp, proj_b, x, ln1_g, ln1_b, yf, ybf);
  // 6) fused MLP + residual + LN2 -> d_out
  k_mlp_ln2<<<NWIN, 256, 0, stream>>>(ybf, yf, w1, fc1_b, w2, fc2_b, ln2_g, ln2_b,
                                      (float*)d_out);
  (void)in_sizes; (void)n_in; (void)out_size; (void)ws_size;
}
